// SingleHeadedAttention_59674275610635
// MI455X (gfx1250) — matrix operand order repaired and checked by reading, not compiled, not run on hardware
//
#include <hip/hip_runtime.h>
#include <stddef.h>

#define DIM   512
#define NSEQ  4096
#define BATCH 8
#define BR    64
#define BC    64
#define NJ    (NSEQ / BC)

// Toolchain arity for __builtin_amdgcn_tensor_load_to_lds:
//   5 = ROCm 7.2 / clang-22, 6 = amdgpu-toolchain / clang-23
// Compiler reported "expected 6" -> this lane is the 6-arg toolchain.
#ifndef TDM_ARITY
#define TDM_ARITY 6
#endif

typedef __attribute__((ext_vector_type(8)))  unsigned short u16x8;
typedef __attribute__((ext_vector_type(16))) unsigned short u16x16;
typedef __attribute__((ext_vector_type(16))) __bf16         v16bf;
typedef __attribute__((ext_vector_type(8)))  float          v8f;
typedef __attribute__((ext_vector_type(4)))  unsigned int   u32x4;
typedef __attribute__((ext_vector_type(8)))  int            i32x8;
typedef __attribute__((ext_vector_type(4)))  int            i32x4;

// LDS layout (dynamic shared, base offset 0)
#define QS_OFF   0
#define KS_OFF0  65536
#define KS_OFF1  131072
#define VS_OFF   196608
#define SS_OFF   262144
#define PS_OFF   278528
#define AL_OFF   286720
#define LL_OFF   286976
#define SMEM_ATTN 287232

__device__ __forceinline__ unsigned short f2bf(float f) {
  unsigned u = __builtin_bit_cast(unsigned, f);
  u += 0x7FFFu + ((u >> 16) & 1u);            // round-to-nearest-even
  return (unsigned short)(u >> 16);
}

__device__ __forceinline__ v8f wmma_bf16(v16bf a, v16bf b, v8f c) {
  return __builtin_amdgcn_wmma_f32_16x16x32_bf16(false, a, false, b, (short)0, c, false, false);
}

// A fragment (16x32, MxK): lane holds row lane&15; K elements at hi*8 .. +7 and +16 .. +23
__device__ __forceinline__ v16bf load_A(const unsigned short* base, int ld) {
  const int lane = threadIdx.x & 31;
  const unsigned short* p = base + (lane & 15) * ld + ((lane >> 4) << 3);
  u16x8 lo = *(const u16x8*)p;
  u16x8 hh = *(const u16x8*)(p + 16);
  u16x16 c = __builtin_shufflevector(lo, hh, 0,1,2,3,4,5,6,7,8,9,10,11,12,13,14,15);
  return __builtin_bit_cast(v16bf, c);
}

// B fragment (32x16, KxN) read from B^T storage: lane holds col lane&15; K at hi*8 .. +7 and +16 .. +23
__device__ __forceinline__ v16bf load_B(const unsigned short* base, int ld) {
  const int lane = threadIdx.x & 31;
  const unsigned short* p = base + (lane & 15) * ld + ((lane >> 4) << 3);
  u16x8 lo = *(const u16x8*)p;
  u16x8 hh = *(const u16x8*)(p + 16);
  u16x16 c = __builtin_shufflevector(lo, hh, 0,1,2,3,4,5,6,7,8,9,10,11,12,13,14,15);
  return __builtin_bit_cast(v16bf, c);
}

// ---- Tensor Data Mover: issue one descriptor (whole tile) ----
// tile_d1 == 0 -> 1-D tile of tile_d0 elements; else 2-D tile_d0 x tile_d1, row stride stride0.
// Element size fixed at 2 bytes (bf16). Huge tensor dims so no OOB clamping occurs.
__device__ __forceinline__ void tdm_load_bf16(unsigned lds_off, const unsigned short* src,
                                              unsigned tile_d0, unsigned tile_d1,
                                              unsigned stride0) {
  unsigned long long ga = (unsigned long long)(size_t)src;
  u32x4 g0;
  g0[0] = 1u;                                            // count=1, user load
  g0[1] = lds_off;                                       // lds_addr (bytes)
  g0[2] = (unsigned)(ga & 0xFFFFFFFFu);                  // global_addr[31:0]
  g0[3] = (unsigned)((ga >> 32) & 0x01FFFFFFu) | (2u << 30); // ga[56:32] | type=2
  i32x8 g1;
  g1[0] = (int)(1u << 16);                               // data_size=1 (2 bytes)
  g1[1] = (int)0xFFFF0000u;                              // tensor_dim0 = 0x7FFFFFFF (lo16<<16)
  g1[2] = (int)(0x7FFFu | (0xFFFFu << 16));              // dim0 hi16 | tensor_dim1 lo16
  g1[3] = (int)(0x7FFFu | (tile_d0 << 16));              // dim1 hi16 | tile_dim0
  g1[4] = (int)(tile_d1 & 0xFFFFu);                      // tile_dim1 (0 => 1-D), tile_dim2=0
  g1[5] = (int)stride0;                                  // tensor_dim0_stride lo32
  g1[6] = 0;
  g1[7] = 0;
  i32x4 z = {0, 0, 0, 0};
#if TDM_ARITY == 6
  i32x8 z8 = {0, 0, 0, 0, 0, 0, 0, 0};
  __builtin_amdgcn_tensor_load_to_lds(g0, g1, z, z, z8, 0);
#else
  __builtin_amdgcn_tensor_load_to_lds(g0, g1, z, z, 0);
#endif
}

// ---------------- Kernel 1: W[in][out] f32 -> Wt[out][in] bf16 ----------------
extern "C" __global__ __launch_bounds__(256) void wtrans_kernel(
    const float* __restrict__ W, unsigned short* __restrict__ Wt) {
  int idx  = blockIdx.x * 256 + threadIdx.x;     // over DIM*DIM
  int dout = idx >> 9;
  int din  = idx & (DIM - 1);
  Wt[idx]  = f2bf(W[din * DIM + dout]);
}

// ---------------- Kernel 2: QKV projection ----------------
extern "C" __global__ __launch_bounds__(256) void qkv_kernel(
    const float* __restrict__ x,
    const unsigned short* __restrict__ WqT, const unsigned short* __restrict__ WkT,
    const unsigned short* __restrict__ WvT,
    const float* __restrict__ bq, const float* __restrict__ bk, const float* __restrict__ bv,
    unsigned short* __restrict__ qb, unsigned short* __restrict__ kb,
    unsigned short* __restrict__ vT) {
  extern __shared__ char smem[];
  unsigned short* Xs = (unsigned short*)smem;    // BR x DIM bf16
  const int b   = blockIdx.y;
  const int i0  = blockIdx.x * BR;
  const int tid = threadIdx.x;
  const int wave = tid >> 5, lane = tid & 31;
  const int r0 = (wave & 3) * 16, chalf = wave >> 2;
  const int col = lane & 15, hi = lane >> 4;

  const float* xg = x + ((size_t)b * NSEQ + i0) * DIM;
  for (int idx = tid; idx < BR * DIM; idx += 256) Xs[idx] = f2bf(xg[idx]);
  __syncthreads();

  // ---- Q, K: row-major bf16 stores ----
  for (int which = 0; which < 2; ++which) {
    const unsigned short* Wt = which ? WkT : WqT;
    const float* bias_p      = which ? bk  : bq;
    unsigned short* dst      = which ? kb  : qb;
    for (int tt = 0; tt < 16; ++tt) {
      const int n0 = chalf * 256 + tt * 16;
      v8f acc = {};
      #pragma unroll
      for (int kk = 0; kk < DIM / 32; ++kk)
        acc = wmma_bf16(load_A(Xs + r0 * DIM + kk * 32, DIM),
                        load_B(Wt + (size_t)n0 * DIM + kk * 32, DIM), acc);
      const float bias = bias_p[n0 + col];
      unsigned short* orow = dst + ((size_t)b * NSEQ + i0 + r0 + 8 * hi) * DIM + n0 + col;
      #pragma unroll
      for (int v = 0; v < 8; ++v) orow[(size_t)v * DIM] = f2bf(acc[v] + bias);
    }
  }

  // ---- V: transposed store vT[b][dout][n] (16B packed) ----
  for (int tt = 0; tt < 16; ++tt) {
    const int n0 = chalf * 256 + tt * 16;
    v8f acc = {};
    #pragma unroll
    for (int kk = 0; kk < DIM / 32; ++kk)
      acc = wmma_bf16(load_A(Xs + r0 * DIM + kk * 32, DIM),
                      load_B(WvT + (size_t)n0 * DIM + kk * 32, DIM), acc);
    const float bias = bv[n0 + col];
    u16x8 pk;
    #pragma unroll
    for (int v = 0; v < 8; ++v) pk[v] = f2bf(acc[v] + bias);
    *(u16x8*)(vT + ((size_t)b * DIM + n0 + col) * NSEQ + i0 + r0 + 8 * hi) = pk;
  }
}

// ---------------- Kernel 3: flash attention + fused output projection ----------------
extern "C" __global__ __launch_bounds__(256) void attn_kernel(
    const unsigned short* __restrict__ qb, const unsigned short* __restrict__ kb,
    const unsigned short* __restrict__ vT, const unsigned short* __restrict__ WoT,
    const float* __restrict__ bo, float* __restrict__ out) {
  extern __shared__ char smem[];
  unsigned short* Qs  = (unsigned short*)(smem + QS_OFF);   // 64x512 bf16
  unsigned short* Ksb[2];
  Ksb[0] = (unsigned short*)(smem + KS_OFF0);               // 64x512 bf16 (double buffered)
  Ksb[1] = (unsigned short*)(smem + KS_OFF1);
  unsigned short* Vs  = (unsigned short*)(smem + VS_OFF);   // 512x64 bf16 (d-major)
  float*          Ss  = (float*)(smem + SS_OFF);            // 64x64 f32
  unsigned short* Ps  = (unsigned short*)(smem + PS_OFF);   // 64x64 bf16
  float*    alphaArr  = (float*)(smem + AL_OFF);            // 64
  float*        lArr  = (float*)(smem + LL_OFF);            // 64

  const int b   = blockIdx.y;
  const int i0  = blockIdx.x * BR;
  const int tid = threadIdx.x;
  const int wave = tid >> 5, lane = tid & 31;
  const int r0 = (wave & 3) * 16, chalf = wave >> 2;
  const int col = lane & 15, hi = lane >> 4;
  const float scale = 0.04419417382415922f;                 // 1/sqrt(512)

  const unsigned short* kgB = kb + (size_t)b * NSEQ * DIM;
  const unsigned short* vgB = vT + (size_t)b * DIM * NSEQ;

  // Kick off TDM for K(0) (one wave issues; EXEC ignored by TDM, one op per wave)
  if (wave == 0) {
    tdm_load_bf16(KS_OFF0, kgB, BC * DIM, 0, BC * DIM);     // 1-D 32768-element tile
  }

  // Q tile -> LDS (overlaps the K(0) DMA)
  const unsigned short* qg = qb + ((size_t)b * NSEQ + i0) * DIM;
  for (int idx = tid; idx < BR * DIM / 8; idx += 256)
    ((u16x8*)Qs)[idx] = ((const u16x8*)qg)[idx];

  v8f oacc[16];
  #pragma unroll
  for (int tt = 0; tt < 16; ++tt) oacc[tt] = (v8f){};
  float m_i = -1e30f, l_i = 0.f;

  for (int j = 0; j < NJ; ++j) {
    const unsigned short* Ks = Ksb[j & 1];
    if (wave == 0) __builtin_amdgcn_s_wait_tensorcnt(0);    // K(j) (and V(j-1)) landed
    __syncthreads();                                        // K(j) visible; Vs free
    if (wave == 0) {
      tdm_load_bf16(VS_OFF, vgB + (size_t)j * BC, BC, DIM, NSEQ);  // V(j): 2-D 64x512
      if (j + 1 < NJ)
        tdm_load_bf16((j & 1) ? KS_OFF0 : KS_OFF1,
                      kgB + (size_t)(j + 1) * BC * DIM, BC * DIM, 0, BC * DIM);
    }

    // S = Q K^T (each wave: 2 tiles of 16x16, K-depth 512) — overlaps V(j)/K(j+1) DMA
    for (int ct = 0; ct < 2; ++ct) {
      const int n0 = chalf * 32 + ct * 16;
      v8f acc = {};
      #pragma unroll
      for (int kk = 0; kk < DIM / 32; ++kk)
        acc = wmma_bf16(load_A(Qs + r0 * DIM + kk * 32, DIM),
                        load_B(Ks + n0 * DIM + kk * 32, DIM), acc);
      #pragma unroll
      for (int v = 0; v < 8; ++v)
        Ss[(r0 + v + 8 * hi) * BC + n0 + col] = acc[v] * scale;
    }
    __syncthreads();

    // online softmax: lane owns row r0+col, cols hi*32..hi*32+31 (halves merged by shfl)
    {
      const int mr = r0 + col;
      float sv[32], pm = -1e30f;
      #pragma unroll
      for (int c = 0; c < 32; ++c) { sv[c] = Ss[mr * BC + hi * 32 + c]; pm = fmaxf(pm, sv[c]); }
      pm = fmaxf(pm, __shfl_xor(pm, 16, 32));
      const float mnew  = fmaxf(m_i, pm);
      const float alpha = __expf(m_i - mnew);
      float psum = 0.f;
      #pragma unroll
      for (int c = 0; c < 32; ++c) {
        float p = __expf(sv[c] - mnew);
        psum += p;
        Ps[mr * BC + hi * 32 + c] = f2bf(p);
      }
      psum += __shfl_xor(psum, 16, 32);
      l_i = l_i * alpha + psum;
      m_i = mnew;
      if (chalf == 0 && hi == 0) alphaArr[mr] = alpha;
    }
    // V(j) is first outstanding tensor op -> wait leaves K(j+1) in flight
    if (wave == 0) {
      if (j + 1 < NJ) __builtin_amdgcn_s_wait_tensorcnt(1);
      else            __builtin_amdgcn_s_wait_tensorcnt(0);
    }
    __syncthreads();

    // rescale O, then O += P V
    float a8[8];
    #pragma unroll
    for (int v = 0; v < 8; ++v) a8[v] = alphaArr[r0 + v + 8 * hi];
    #pragma unroll
    for (int tt = 0; tt < 16; ++tt) {
      #pragma unroll
      for (int v = 0; v < 8; ++v) oacc[tt][v] *= a8[v];
    }
    #pragma unroll
    for (int tt = 0; tt < 16; ++tt) {
      const int d0 = chalf * 256 + tt * 16;
      #pragma unroll
      for (int ks = 0; ks < 2; ++ks)
        oacc[tt] = wmma_bf16(load_A(Ps + r0 * BC + ks * 32, BC),
                             load_B(Vs + d0 * BC + ks * 32, BC), oacc[tt]);
    }
  }

  // normalize by row sums, stage O in LDS (bf16, reuse Qs), fused out-proj
  if (chalf == 0 && hi == 0) lArr[r0 + col] = l_i;
  __syncthreads();
  float li8[8];
  #pragma unroll
  for (int v = 0; v < 8; ++v) li8[v] = 1.0f / lArr[r0 + v + 8 * hi];
  unsigned short* Os = Qs;
  #pragma unroll
  for (int tt = 0; tt < 16; ++tt) {
    const int d0 = chalf * 256 + tt * 16;
    #pragma unroll
    for (int v = 0; v < 8; ++v)
      Os[(r0 + v + 8 * hi) * DIM + d0 + col] = f2bf(oacc[tt][v] * li8[v]);
  }
  __syncthreads();

  float* outg = out + ((size_t)b * NSEQ + i0) * DIM;
  for (int tt = 0; tt < 16; ++tt) {
    const int n0 = chalf * 256 + tt * 16;
    v8f f = {};
    #pragma unroll
    for (int kk = 0; kk < DIM / 32; ++kk)
      f = wmma_bf16(load_A(Os + r0 * DIM + kk * 32, DIM),
                    load_B(WoT + (size_t)n0 * DIM + kk * 32, DIM), f);
    const float bias = bo[n0 + col];
    #pragma unroll
    for (int v = 0; v < 8; ++v)
      outg[(size_t)(r0 + v + 8 * hi) * DIM + n0 + col] = f[v] + bias;
  }
}

// ---------------- host launch ----------------
extern "C" void kernel_launch(void* const* d_in, const int* in_sizes, int n_in,
                              void* d_out, int out_size, void* d_ws, size_t ws_size,
                              hipStream_t stream) {
  (void)in_sizes; (void)n_in; (void)out_size; (void)ws_size;
  const float* x  = (const float*)d_in[0];
  const float* Wq = (const float*)d_in[1];
  const float* bq = (const float*)d_in[2];
  const float* Wk = (const float*)d_in[3];
  const float* bk = (const float*)d_in[4];
  const float* Wv = (const float*)d_in[5];
  const float* bv = (const float*)d_in[6];
  const float* Wo = (const float*)d_in[7];
  const float* bo = (const float*)d_in[8];
  float* out = (float*)d_out;

  unsigned short* WqT = (unsigned short*)d_ws;
  unsigned short* WkT = WqT + (size_t)DIM * DIM;
  unsigned short* WvT = WkT + (size_t)DIM * DIM;
  unsigned short* WoT = WvT + (size_t)DIM * DIM;
  unsigned short* qb  = WoT + (size_t)DIM * DIM;
  unsigned short* kb  = qb + (size_t)BATCH * NSEQ * DIM;
  unsigned short* vT  = kb + (size_t)BATCH * NSEQ * DIM;

  wtrans_kernel<<<DIM * DIM / 256, 256, 0, stream>>>(Wq, WqT);
  wtrans_kernel<<<DIM * DIM / 256, 256, 0, stream>>>(Wk, WkT);
  wtrans_kernel<<<DIM * DIM / 256, 256, 0, stream>>>(Wv, WvT);
  wtrans_kernel<<<DIM * DIM / 256, 256, 0, stream>>>(Wo, WoT);

  dim3 grid(NSEQ / BR, BATCH);
  qkv_kernel<<<grid, 256, BR * DIM * 2, stream>>>(x, WqT, WkT, WvT, bq, bk, bv, qb, kb, vT);
  attn_kernel<<<grid, 256, SMEM_ATTN, stream>>>(qb, kb, vT, WoT, bo, out);
}
